// AutoRegressive_88381837017909
// MI455X (gfx1250) — compile-verified, matrix operand order checked by reading, not run on hardware
//
#include <hip/hip_runtime.h>
#include <hip/hip_bf16.h>

typedef __attribute__((ext_vector_type(16))) __bf16 v16bf;
typedef __attribute__((ext_vector_type(8)))  float  v8f;

#define UNITS  768
#define IND    96
#define NGATE  3072          // 4*UNITS
#define KTOT   864           // IND + UNITS  (= 27*32)
#define TSTEPS 64
#define SEQ    24
#define MROWS  32            // batch rows per workgroup (2 WMMA m-tiles)
#define SS     880           // padded LDS row stride (bf16 elements), 1760B = 55*32B

__device__ __forceinline__ unsigned short f2bf(float f) {
    unsigned u = __float_as_uint(f);
    u = (u + 0x7FFFu + ((u >> 16) & 1u)) >> 16;   // round-to-nearest-even
    return (unsigned short)u;
}
__device__ __forceinline__ float sigmoidf(float x) { return 1.0f / (1.0f + __expf(-x)); }
__device__ __forceinline__ float tanh_fast(float x) {
    float e = __expf(2.0f * x);
    return (e - 1.0f) / (e + 1.0f);
}
__device__ __forceinline__ v8f splat8(float b) { v8f v = {b,b,b,b,b,b,b,b}; return v; }

// ---- one-time prep: weights -> bf16, fused bias -----------------------------
__global__ void cvt_kernel(const float* __restrict__ Wih, const float* __restrict__ Whh,
                           const float* __restrict__ bih, const float* __restrict__ bhh,
                           const float* __restrict__ Wd,
                           unsigned short* __restrict__ wc, unsigned short* __restrict__ wd,
                           float* __restrict__ bsum) {
    int i = blockIdx.x * 256 + threadIdx.x;
    const int totWc = NGATE * KTOT;
    const int totWd = IND * UNITS;
    if (i < totWc) {
        int n = i / KTOT, k = i - n * KTOT;
        float v = (k < IND) ? Wih[n * IND + k] : Whh[n * UNITS + (k - IND)];
        wc[i] = f2bf(v);
    } else if (i < totWc + totWd) {
        int j = i - totWc;
        wd[j] = f2bf(Wd[j]);
    } else if (i < totWc + totWd + NGATE) {
        int n = i - totWc - totWd;
        bsum[n] = bih[n] + bhh[n];
    }
}

// ---- persistent autoregressive LSTM kernel ---------------------------------
// grid = B/32 blocks, 256 threads (8 waves), exactly one WG per WGP
// (launch_bounds(256,1) -> full VGPR budget, no spills). Each block runs all
// 64 steps for its 32 batch rows. State [x(96)|h(768)] double-buffered bf16 in
// LDS; cell state c lives in VGPRs. Weights stream from L2 each step.
__global__ void __launch_bounds__(256, 1)
lstm_ar_kernel(const float* __restrict__ inputs,
               const unsigned short* __restrict__ wc,   // [3072][864] bf16
               const unsigned short* __restrict__ wd,   // [96][768]  bf16
               const float* __restrict__ bsum,          // [3072] b_ih+b_hh
               const float* __restrict__ bd,            // [96]
               float* __restrict__ out) {               // [B][64][96]
    extern __shared__ __align__(64) unsigned short Sm[];   // 2 * MROWS * SS
    const int wave    = threadIdx.x >> 5;
    const int lane    = threadIdx.x & 31;
    const int lane_n  = lane & 15;
    const int lane_hi = lane >> 4;     // 0: k 0..15 / m 0..7 ; 1: k 16..31 / m 8..15
    const int b0      = blockIdx.x * MROWS;

    // init: S[0].x = inputs[:, SEQ-1, :], S[0].h = 0, S[1] = 0
    for (int idx = threadIdx.x; idx < MROWS * SS; idx += blockDim.x) {
        int row = idx / SS, k = idx - row * SS;
        unsigned short v = 0;
        if (k < IND) v = f2bf(inputs[(size_t)(b0 + row) * SEQ * IND + (SEQ - 1) * IND + k]);
        Sm[idx] = v;
        Sm[MROWS * SS + idx] = 0;
    }
    float creg[6][2][8];
    #pragma unroll
    for (int a = 0; a < 6; ++a)
        #pragma unroll
        for (int m = 0; m < 2; ++m)
            #pragma unroll
            for (int r = 0; r < 8; ++r) creg[a][m][r] = 0.0f;
    __syncthreads();

    int cur = 0;
    for (int t = 0; t < TSTEPS; ++t) {
        const int nxt = cur ^ 1;
        const unsigned short* Sc = Sm + cur * MROWS * SS;
        unsigned short*       Sn = Sm + nxt * MROWS * SS;

        // ---- phase A: gates = [x|h] @ Wc^T + b ; LSTM elementwise ; h -> Sn.h
        #pragma unroll
        for (int ut = 0; ut < 6; ++ut) {
            const int nunit = wave * 96 + ut * 16 + lane_n;   // this lane's unit column
            v8f acc[4][2];
            #pragma unroll
            for (int g = 0; g < 4; ++g) {
                float b = bsum[g * UNITS + nunit];
                acc[g][0] = splat8(b);
                acc[g][1] = splat8(b);
            }
            const unsigned short* aptr0 = Sc + (lane_n) * SS + lane_hi * 16;
            const unsigned short* aptr1 = Sc + (16 + lane_n) * SS + lane_hi * 16;
            const unsigned short* bptr0 = wc + (size_t)(0 * UNITS + nunit) * KTOT + lane_hi * 16;
            const unsigned short* bptr1 = wc + (size_t)(1 * UNITS + nunit) * KTOT + lane_hi * 16;
            const unsigned short* bptr2 = wc + (size_t)(2 * UNITS + nunit) * KTOT + lane_hi * 16;
            const unsigned short* bptr3 = wc + (size_t)(3 * UNITS + nunit) * KTOT + lane_hi * 16;
            #pragma unroll 1
            for (int kc = 0; kc < KTOT / 32; ++kc) {
                const int k0 = kc * 32;
                __builtin_prefetch(bptr0 + k0 + 64, 0, 3);   // WGP-scope global_prefetch_b8
                __builtin_prefetch(bptr2 + k0 + 64, 0, 3);
                v16bf a0 = *(const v16bf*)(aptr0 + k0);
                v16bf a1 = *(const v16bf*)(aptr1 + k0);
                v16bf w0 = *(const v16bf*)(bptr0 + k0);
                v16bf w1 = *(const v16bf*)(bptr1 + k0);
                v16bf w2 = *(const v16bf*)(bptr2 + k0);
                v16bf w3 = *(const v16bf*)(bptr3 + k0);
                acc[0][0] = __builtin_amdgcn_wmma_f32_16x16x32_bf16(false, a0, false, w0, (short)0, acc[0][0], false, false);
                acc[0][1] = __builtin_amdgcn_wmma_f32_16x16x32_bf16(false, a1, false, w0, (short)0, acc[0][1], false, false);
                acc[1][0] = __builtin_amdgcn_wmma_f32_16x16x32_bf16(false, a0, false, w1, (short)0, acc[1][0], false, false);
                acc[1][1] = __builtin_amdgcn_wmma_f32_16x16x32_bf16(false, a1, false, w1, (short)0, acc[1][1], false, false);
                acc[2][0] = __builtin_amdgcn_wmma_f32_16x16x32_bf16(false, a0, false, w2, (short)0, acc[2][0], false, false);
                acc[2][1] = __builtin_amdgcn_wmma_f32_16x16x32_bf16(false, a1, false, w2, (short)0, acc[2][1], false, false);
                acc[3][0] = __builtin_amdgcn_wmma_f32_16x16x32_bf16(false, a0, false, w3, (short)0, acc[3][0], false, false);
                acc[3][1] = __builtin_amdgcn_wmma_f32_16x16x32_bf16(false, a1, false, w3, (short)0, acc[3][1], false, false);
            }
            // LSTM elementwise; accumulator element (m,n): m = mt*16 + lane_hi*8 + r, n = lane_n
            #pragma unroll
            for (int mt = 0; mt < 2; ++mt) {
                #pragma unroll
                for (int r = 0; r < 8; ++r) {
                    float iv = sigmoidf(acc[0][mt][r]);
                    float fv = sigmoidf(acc[1][mt][r]);
                    float gv = tanh_fast(acc[2][mt][r]);
                    float ov = sigmoidf(acc[3][mt][r]);
                    float c  = fv * creg[ut][mt][r] + iv * gv;
                    creg[ut][mt][r] = c;
                    float h  = ov * tanh_fast(c);
                    int mrow = mt * 16 + lane_hi * 8 + r;
                    Sn[mrow * SS + IND + nunit] = f2bf(h);
                }
            }
        }
        __syncthreads();

        // ---- phase B: pred = h_new @ Wd^T + b_d -> out[:,t,:] and Sn.x
        if (wave < 6) {
            const int ncol = wave * 16 + lane_n;
            float bb = bd[ncol];
            v8f pacc0 = splat8(bb);
            v8f pacc1 = splat8(bb);
            const unsigned short* aptr0 = Sn + (lane_n) * SS + IND + lane_hi * 16;
            const unsigned short* aptr1 = Sn + (16 + lane_n) * SS + IND + lane_hi * 16;
            const unsigned short* bptr  = wd + (size_t)ncol * UNITS + lane_hi * 16;
            #pragma unroll 1
            for (int kc = 0; kc < UNITS / 32; ++kc) {
                const int k0 = kc * 32;
                v16bf a0 = *(const v16bf*)(aptr0 + k0);
                v16bf a1 = *(const v16bf*)(aptr1 + k0);
                v16bf wm = *(const v16bf*)(bptr + k0);
                pacc0 = __builtin_amdgcn_wmma_f32_16x16x32_bf16(false, a0, false, wm, (short)0, pacc0, false, false);
                pacc1 = __builtin_amdgcn_wmma_f32_16x16x32_bf16(false, a1, false, wm, (short)0, pacc1, false, false);
            }
            #pragma unroll
            for (int mt = 0; mt < 2; ++mt) {
                #pragma unroll
                for (int r = 0; r < 8; ++r) {
                    int mrow = mt * 16 + lane_hi * 8 + r;
                    float p = (mt == 0) ? pacc0[r] : pacc1[r];
                    out[(size_t)(b0 + mrow) * TSTEPS * IND + (size_t)t * IND + ncol] = p;
                    Sn[mrow * SS + ncol] = f2bf(p);
                }
            }
        }
        __syncthreads();
        cur = nxt;
    }
}

extern "C" void kernel_launch(void* const* d_in, const int* in_sizes, int n_in,
                              void* d_out, int out_size, void* d_ws, size_t ws_size,
                              hipStream_t stream) {
    const float* inputs = (const float*)d_in[0];
    const float* W_ih   = (const float*)d_in[1];
    const float* W_hh   = (const float*)d_in[2];
    const float* b_ih   = (const float*)d_in[3];
    const float* b_hh   = (const float*)d_in[4];
    const float* W_d    = (const float*)d_in[5];
    const float* b_d    = (const float*)d_in[6];

    // workspace layout: bf16 Wc[3072][864] | bf16 Wd[96][768] | f32 bsum[3072]
    unsigned short* wc   = (unsigned short*)d_ws;
    unsigned short* wdp  = wc + (size_t)NGATE * KTOT;
    float*          bsum = (float*)(wdp + (size_t)IND * UNITS);

    const int B = in_sizes[0] / (SEQ * IND);   // 2048

    const int total = NGATE * KTOT + IND * UNITS + NGATE;
    cvt_kernel<<<(total + 255) / 256, 256, 0, stream>>>(W_ih, W_hh, b_ih, b_hh, W_d, wc, wdp, bsum);

    const size_t shmem = (size_t)2 * MROWS * SS * sizeof(unsigned short);  // ~110 KB < 320 KB/WGP
    lstm_ar_kernel<<<B / MROWS, 256, shmem, stream>>>(inputs, wc, wdp, bsum, b_d, (float*)d_out);
}
